// LSTMStateBufferCell_39376260169764
// MI455X (gfx1250) — compile-verified, matrix operand order checked by reading, not run on hardware
//
#include <hip/hip_runtime.h>
#include <hip/hip_bf16.h>

// Shapes fixed by the reference: S=1024, B=64, H=1024, fp32 data, int32 masks/op.
#define S_DIM 1024
#define B_DIM 64
#define H_DIM 1024

typedef __attribute__((ext_vector_type(8))) int v8i;

// ---------------------------------------------------------------------------
// Kernel A: pos[b] = sum_s masks[s][b]  via  ones(16x64,u8) x maskTile(64x16,u8)
// with V_WMMA_I32_16X16X64_IU8.
//
// Parallel decomposition:
//   grid.x = 4 blocks, block g owns batch columns [g*16, g*16+16)
//   4 waves per block, wave w owns S-slice [w*256, w*256+256) -> 4 WMMA accums
//   partial column sums combined through LDS + workgroup barrier.
// Since A is all-ones, every row of D equals the column sums, so each lane's
// acc[0] holds the partial pos for its column.
// ---------------------------------------------------------------------------
__global__ void __launch_bounds__(128)
lstm_pos_wmma_kernel(const int* __restrict__ masks, int* __restrict__ pos_out) {
    __shared__ int partial[4][16];

    const int lane  = threadIdx.x & 31;
    const int slice = threadIdx.x >> 5;            // 0..3 : S-slice this wave owns
    const int col   = blockIdx.x * 16 + (lane & 15); // batch column this lane owns
    const int khi   = lane & 16;                   // lanes 16-31 hold K+16 halves

    // A matrix: 16x64 u8 all-ones -> every byte = 0x01 in all 8 VGPRs.
    v8i amat;
#pragma unroll
    for (int i = 0; i < 8; ++i) amat[i] = 0x01010101;

    v8i acc = {0, 0, 0, 0, 0, 0, 0, 0};

    // This wave accumulates its 256-row slice in 4 chunks of K=64.
    const int sbase = slice * 256;
#pragma unroll
    for (int sc = 0; sc < 4; ++sc) {
        const int s0 = sbase + sc * 64;
        v8i bmat;
#pragma unroll
        for (int v = 0; v < 8; ++v) {
            // 8-bit B-matrix (64x16) VGPR layout:
            //   V0..V3: lanes 0-15 -> K=0..15,  lanes 16-31 -> K=16..31
            //   V4..V7: lanes 0-15 -> K=32..47, lanes 16-31 -> K=48..63
            const int base_k = ((v < 4) ? 0 : 32) + khi + (v & 3) * 4;
            int packed = 0;
#pragma unroll
            for (int j = 0; j < 4; ++j) {
                const int m = masks[(s0 + base_k + j) * B_DIM + col]; // 0 or 1
                packed |= (m & 0xFF) << (8 * j);
            }
            bmat[v] = packed;
        }
        // (sgn_a, A, sgn_b, B, C, reuse_a, reuse_b) — unsigned IU8, i32 accum
        acc = __builtin_amdgcn_wmma_i32_16x16x64_iu8(false, amat, false, bmat,
                                                     acc, false, false);
    }

    // All 16 D rows are identical (A all-ones); lanes 0-15 carry the 16
    // columns of this block in acc[0]. Publish this wave's partial sum.
    if (lane < 16) partial[slice][lane] = acc[0];
    __syncthreads();

    // Combine the 4 S-slices and store pos for this block's 16 columns.
    if (threadIdx.x < 16) {
        const int c = threadIdx.x;
        pos_out[blockIdx.x * 16 + c] =
            partial[0][c] + partial[1][c] + partial[2][c] + partial[3][c];
    }
}

// ---------------------------------------------------------------------------
// Kernel B: select + gather. a=|op[b]| is exactly 0 or 1, so the blend
// prev*a + cur*(1-a) is an exact select: fetch only one source row per
// output. sel==0 maps to the broadcast init vector; sel>=1 maps to row sel-1
// of hiddens/cells. float4 => global_load_b128 / global_store_b128, coalesced.
// Grid: blockIdx.x = b*2 + which  (which: 0=hidden, 1=cell), 256 thr/block,
// each thread moves one float4 of the H=1024 row.
// ---------------------------------------------------------------------------
__global__ void __launch_bounds__(256)
lstm_gather_kernel(const float* __restrict__ hiddens,
                   const float* __restrict__ cells,
                   const float* __restrict__ init_hidden,
                   const float* __restrict__ init_cell,
                   const int*   __restrict__ op,
                   const int*   __restrict__ pos_buf,
                   float* __restrict__ out) {
    const int b     = blockIdx.x >> 1;
    const int which = blockIdx.x & 1;
    const int t     = threadIdx.x;          // float4 index within the row

    const int pos = pos_buf[b];
    const int o   = op[b];
    const int a   = (o < 0) ? -o : o;       // 0 or 1
    int prev = pos - 1;
    if (prev < 0) prev += S_DIM + 1;        // mod (S+1)
    const int sel = a ? prev : pos;

    const float* src = which ? cells      : hiddens;
    const float* ini = which ? init_cell  : init_hidden;

    const float4* srcv;
    if (sel == 0) {
        srcv = (const float4*)ini;
    } else {
        srcv = (const float4*)(src + (size_t)(sel - 1) * B_DIM * H_DIM
                                   + (size_t)b * H_DIM);
    }
    const float4 v = srcv[t];

    float4* outv = (float4*)(out + (size_t)which * B_DIM * H_DIM
                                 + (size_t)b * H_DIM);
    outv[t] = v;
}

// ---------------------------------------------------------------------------
// Launch. Inputs (setup_inputs order):
//   0 hiddens (S,B,H) f32   1 cells (S,B,H) f32
//   2 init_hidden (H,) f32  3 init_cell (H,) f32
//   4 hidden_masks (S,B) i32  5 op (B,) i32
// Output: hidden_ret (B,H) then cell_ret (B,H), fp32, concatenated.
// d_ws: first 64 ints hold pos[b].
// ---------------------------------------------------------------------------
extern "C" void kernel_launch(void* const* d_in, const int* in_sizes, int n_in,
                              void* d_out, int out_size, void* d_ws, size_t ws_size,
                              hipStream_t stream) {
    const float* hiddens     = (const float*)d_in[0];
    const float* cells       = (const float*)d_in[1];
    const float* init_hidden = (const float*)d_in[2];
    const float* init_cell   = (const float*)d_in[3];
    const int*   masks       = (const int*)d_in[4];
    const int*   op          = (const int*)d_in[5];
    float*       out         = (float*)d_out;
    int*         pos_buf     = (int*)d_ws;

    // Kernel A: 4 blocks x 4 waves; WMMA column-sum of masks, S split 4-way.
    lstm_pos_wmma_kernel<<<4, 128, 0, stream>>>(masks, pos_buf);

    // Kernel B: 64 batches x {hidden, cell} = 128 blocks of 256 threads.
    lstm_gather_kernel<<<B_DIM * 2, 256, 0, stream>>>(
        hiddens, cells, init_hidden, init_cell, op, pos_buf, out);
}
